// FlatSquare7x7NNUEv4_47519518163400
// MI455X (gfx1250) — matrix operand order repaired and checked by reading, not compiled, not run on hardware
//
#include <hip/hip_runtime.h>

typedef _Float16 half_t;
typedef __attribute__((ext_vector_type(16))) _Float16 v16h;
typedef __attribute__((ext_vector_type(8)))  _Float16 v8h;
typedef __attribute__((ext_vector_type(8)))  float    v8f;

#define QMAXF (127.0f/128.0f)

// 16 chunks in group order: groups 0(4), 1(8), 2(4)
__constant__ int c_y0[16] = {0,0,3,3, 0,1,3,2,1,0,2,3, 1,1,2,2};
__constant__ int c_x0[16] = {0,3,3,0, 1,3,2,0,0,2,3,1, 1,2,2,1};
__constant__ int c_rk[16] = {0,1,2,3, 0,1,2,3,0,1,2,3, 0,1,2,3};
__constant__ int c_tr[16] = {0,0,0,0, 0,0,0,0,1,1,1,1, 0,0,0,0};
__constant__ int c_gr[16] = {0,0,0,0, 1,1,1,1,1,1,1,1, 2,2,2,2};

// workspace half-region offsets (in halves, from d_ws)
#define OFF_W0   0         // [3][128][32]  (K padded 8->32)
#define OFF_W1   12288     // [3][128][128]
#define OFF_W2   61440     // [3][128][512] (K = pos*128+ic)
#define OFF_W3   258048    // [3][128][128]
#define OFF_W4   307200    // [3][128][512]
#define OFF_W5   503808    // [3][32][128]
#define OFF_VW0  516096    // [32][32]
#define OFF_VW1  517120    // [32][32]
#define OFF_VW2  518144    // [16][32] (rows 3..15 zero)
// workspace float-region offsets (floats, region starts at byte 1<<20)
#define FB0 0
#define FB1 384
#define FB2 768
#define FB3 1152
#define FB4 1536
#define FB5 1920
#define FVB0 2016
#define FVB1 2048
#define FVB2 2080
#define FFEAT 4096         // feature accumulator [B][32] f32

#define WMMA(a,b,c) __builtin_amdgcn_wmma_f32_16x16x32_f16(false,(a),false,(b),(short)0,(c),false,false)

// A fragment (16x32 f16) from row-major [row][ld] halves.
// lane<16: halves0-15=K0-15 ; lane>=16: K16-31.
__device__ inline v16h load_frag_A(const half_t* base, int ld) {
  int lane = threadIdx.x & 31;
  const half_t* p = base + (lane & 15) * ld + ((lane >> 4) << 4);
  v8h lo = *(const v8h*)p;
  v8h hi = *(const v8h*)(p + 8);
  v16h r;
#pragma unroll
  for (int i = 0; i < 8; ++i) { r[i] = lo[i]; r[i + 8] = hi[i]; }
  return r;
}

// B fragment (32x16 f16) from row-major weights [out][ld] halves.
// lane n (0-15): K 0..15 contiguous ; lane n+16: K 16..31.
__device__ inline v16h load_frag_B(const half_t* base, int ld) {
  int lane = threadIdx.x & 31;
  return *(const v16h*)(base + (lane & 15) * ld + ((lane >> 4) << 4));
}

// D epilogue: bias (per-N), optional relu, f16 store to [samp][128] activation rows
__device__ inline void store_act(const v8f& acc, const float* bias, int nb,
                                 half_t* row0, bool relu) {
  int lane = threadIdx.x & 31;
  int n = nb * 16 + (lane & 15);
  float bv = bias[n];
  half_t* dst = row0 + ((lane >> 4) * 8) * 128 + n;
#pragma unroll
  for (int r = 0; r < 8; ++r) {
    float v = acc[r] + bv;
    if (relu) v = fmaxf(v, 0.f);
    dst[r * 128] = (half_t)v;
  }
}

__device__ inline float bn_scale(const float* bn, int C, int g, int o) {
  const float* p = bn + g * 4 * C;
  return p[o] * rsqrtf(p[3 * C + o] + 1e-5f);
}
__device__ inline float bn_shift(const float* bn, const float* bconv, int C, int g, int o) {
  const float* p = bn + g * 4 * C;
  float s = p[o] * rsqrtf(p[3 * C + o] + 1e-5f);
  return (bconv[g * C + o] - p[2 * C + o]) * s + p[C + o];
}
__device__ inline float q8(float w, float s) {
  float q = rintf(w * s);
  q = fminf(fmaxf(q, -128.f), 127.f);
  return q / s;
}

// ---------------- prep: fold BN into f16 weights, quantize MLP weights ----------------
__global__ __launch_bounds__(256) void prep_kernel(
    const float* w0, const float* b0, const float* w1, const float* b1,
    const float* w2, const float* b2, const float* w3, const float* b3,
    const float* w4, const float* b4, const float* w5, const float* b5,
    const float* bn1, const float* bn2, const float* bn3, const float* bn4, const float* bn5,
    const float* vw0, const float* vb0, const float* vw1, const float* vb1,
    const float* vw2, const float* vb2,
    half_t* wsH, float* wsF)
{
  int tid = blockIdx.x * blockDim.x + threadIdx.x;
  int stride = gridDim.x * blockDim.x;

  // L0: [3][128][2][2][2] -> [3][128][32], K = c*4+dy*2+dx, padded
  for (int i = tid; i < 3*128*32; i += stride) {
    int g = i >> 12, r = i & 4095, o = r >> 5, k = r & 31;
    wsH[OFF_W0 + i] = (half_t)(k < 8 ? w0[(g*128 + o)*8 + k] : 0.f);
  }
  for (int i = tid; i < 384; i += stride) wsF[FB0 + i] = b0[i];

  for (int i = tid; i < 3*128*128; i += stride) {
    int g = i >> 14, r = i & 16383, o = r >> 7;
    wsH[OFF_W1 + i] = (half_t)(w1[i] * bn_scale(bn1, 128, g, o));
  }
  for (int i = tid; i < 384; i += stride) { int g = i >> 7, o = i & 127; wsF[FB1 + i] = bn_shift(bn1, b1, 128, g, o); }

  // L2: [3][128][128][2][2] -> [3][128][512], K = (dy*2+dx)*128 + ic
  for (int i = tid; i < 3*128*512; i += stride) {
    int g = i >> 16, r = i & 65535, o = r >> 9, kk = r & 511;
    int q = kk >> 7, ic = kk & 127;
    wsH[OFF_W2 + i] = (half_t)(w2[(((g*128 + o)*128 + ic) << 2) + q] * bn_scale(bn2, 128, g, o));
  }
  for (int i = tid; i < 384; i += stride) { int g = i >> 7, o = i & 127; wsF[FB2 + i] = bn_shift(bn2, b2, 128, g, o); }

  for (int i = tid; i < 3*128*128; i += stride) {
    int g = i >> 14, r = i & 16383, o = r >> 7;
    wsH[OFF_W3 + i] = (half_t)(w3[i] * bn_scale(bn3, 128, g, o));
  }
  for (int i = tid; i < 384; i += stride) { int g = i >> 7, o = i & 127; wsF[FB3 + i] = bn_shift(bn3, b3, 128, g, o); }

  for (int i = tid; i < 3*128*512; i += stride) {
    int g = i >> 16, r = i & 65535, o = r >> 9, kk = r & 511;
    int q = kk >> 7, ic = kk & 127;
    wsH[OFF_W4 + i] = (half_t)(w4[(((g*128 + o)*128 + ic) << 2) + q] * bn_scale(bn4, 128, g, o));
  }
  for (int i = tid; i < 384; i += stride) { int g = i >> 7, o = i & 127; wsF[FB4 + i] = bn_shift(bn4, b4, 128, g, o); }

  for (int i = tid; i < 3*32*128; i += stride) {
    int g = i >> 12, r = i & 4095, o = r >> 7;
    wsH[OFF_W5 + i] = (half_t)(w5[i] * bn_scale(bn5, 32, g, o));
  }
  for (int i = tid; i < 96; i += stride) { int g = i >> 5, o = i & 31; wsF[FB5 + i] = bn_shift(bn5, b5, 32, g, o); }

  // quantized MLP weights (fake_quant is exact in f16: multiples of 1/256, 1/128)
  for (int i = tid; i < 1024; i += stride) wsH[OFF_VW0 + i] = (half_t)q8(vw0[i], 256.f);
  for (int i = tid; i < 32; i += stride)   wsF[FVB0 + i] = rintf(vb0[i] * 32768.f) * (1.f/32768.f);
  for (int i = tid; i < 1024; i += stride) wsH[OFF_VW1 + i] = (half_t)q8(vw1[i], 128.f);
  for (int i = tid; i < 32; i += stride)   wsF[FVB1 + i] = rintf(vb1[i] * 16384.f) * (1.f/16384.f);
  for (int i = tid; i < 512; i += stride) {
    int o = i >> 5, k = i & 31;
    wsH[OFF_VW2 + i] = (half_t)(o < 3 ? q8(vw2[o*32 + k], 128.f) : 0.f);
  }
  for (int i = tid; i < 16; i += stride)   wsF[FVB2 + i] = (i < 3 ? rintf(vb2[i] * 16384.f) * (1.f/16384.f) : 0.f);
}

// ---------------- conv stack: whole network per 16 patch-samples, all in LDS ----------------
// 8 waves; each wave owns output-channel tile nb == waveId (scalar) for every layer.
__global__ __launch_bounds__(256) void conv_stack_kernel(
    const float* __restrict__ x, const half_t* __restrict__ wsH,
    const float* __restrict__ wsF, float* __restrict__ featAcc, int Bsz)
{
  extern __shared__ char lds_raw[];
  half_t* bufA = (half_t*)lds_raw;             // 18432 halves
  half_t* bufB = (half_t*)(lds_raw + 36864);   // 18432 halves

  const int tid = threadIdx.x;
  // scalar wave id: keeps all job loops wave-uniform (s_cbranch), EXEC all-1s at WMMA
  const int nb = __builtin_amdgcn_readfirstlane(tid) >> 5;
  const int lane = tid & 31;
  const int chunk = (blockIdx.x * 16) / Bsz;
  const int sBase = (blockIdx.x * 16) % Bsz;
  const int g = c_gr[chunk];

  // build L0 A-matrix in bufB: [9 pos][16 samp][32 K], K = c*4+dy*2+dx (8 real, 24 zero)
  {
    const int y0 = c_y0[chunk], x0 = c_x0[chunk], rk = c_rk[chunk], tr = c_tr[chunk];
    for (int idx = tid; idx < 9*16*32; idx += 256) {
      int p = idx >> 9, s = (idx >> 5) & 15, k = idx & 31;
      float val = 0.f;
      if (k < 8) {
        int c = k >> 2, dy = (k >> 1) & 1, dx = k & 1;
        int i = p / 3 + dy, j = p % 3 + dx;     // chunk coords
        int ip = tr ? j : i, jp = tr ? i : j;   // undo transpose
        int sy, sx;                             // undo rot90^k (numpy: R(S)[i][j]=S[j][3-i])
        if      (rk == 0) { sy = ip;     sx = jp;     }
        else if (rk == 1) { sy = jp;     sx = 3 - ip; }
        else if (rk == 2) { sy = 3 - ip; sx = 3 - jp; }
        else              { sy = 3 - jp; sx = ip;     }
        val = x[(((sBase + s) * 2 + c) * 7 + (y0 + sy)) * 7 + (x0 + sx)];
      }
      bufB[idx] = (half_t)val;
    }
  }
  __syncthreads();

  // L0: [144x32]@[32x128] -> act1 (bufA) [9][16][128], relu  (B frag loop-invariant)
  {
    const float* bias = wsF + FB0 + g * 128;
    v16h b = load_frag_B(wsH + OFF_W0 + g * 4096 + nb * 16 * 32, 32);
    for (int p = 0; p < 9; ++p) {
      v16h a = load_frag_A(bufB + p * 512, 32);
      v8f acc = {0,0,0,0,0,0,0,0};
      acc = WMMA(a, b, acc);
      store_act(acc, bias, nb, bufA + p * 2048, true);
    }
  }
  __syncthreads();

  // L1 (1x1, K=128): act1 -> act2 (bufB), relu  (4 B frags hoisted)
  {
    const half_t* W = wsH + OFF_W1 + g * 16384 + nb * 16 * 128;
    const float* bias = wsF + FB1 + g * 128;
    v16h b0 = load_frag_B(W +  0, 128);
    v16h b1 = load_frag_B(W + 32, 128);
    v16h b2 = load_frag_B(W + 64, 128);
    v16h b3 = load_frag_B(W + 96, 128);
    for (int p = 0; p < 9; ++p) {
      const half_t* A = bufA + p * 2048;
      v8f acc = {0,0,0,0,0,0,0,0};
      acc = WMMA(load_frag_A(A +  0, 128), b0, acc);
      acc = WMMA(load_frag_A(A + 32, 128), b1, acc);
      acc = WMMA(load_frag_A(A + 64, 128), b2, acc);
      acc = WMMA(load_frag_A(A + 96, 128), b3, acc);
      store_act(acc, bias, nb, bufB + p * 2048, true);
    }
  }
  __syncthreads();

  // L2 (2x2, K=512): act2 -> act3 (bufA) [4][16][128], relu
  {
    const half_t* W = wsH + OFF_W2 + g * 65536 + nb * 16 * 512;
    const float* bias = wsF + FB2 + g * 128;
    for (int P = 0; P < 4; ++P) {
      int oi = P >> 1, oj = P & 1;
      v8f acc = {0,0,0,0,0,0,0,0};
#pragma unroll
      for (int q = 0; q < 4; ++q) {
        int ip = (oi + (q >> 1)) * 3 + (oj + (q & 1));
#pragma unroll
        for (int kc = 0; kc < 4; ++kc) {
          v16h a = load_frag_A(bufB + ip * 2048 + kc * 32, 128);
          v16h b = load_frag_B(W + (q * 4 + kc) * 32, 512);
          acc = WMMA(a, b, acc);
        }
      }
      store_act(acc, bias, nb, bufA + P * 2048, true);
    }
  }
  __syncthreads();

  // L3 (1x1): act3 -> act4 (bufB), relu  (4 B frags hoisted)
  {
    const half_t* W = wsH + OFF_W3 + g * 16384 + nb * 16 * 128;
    const float* bias = wsF + FB3 + g * 128;
    v16h b0 = load_frag_B(W +  0, 128);
    v16h b1 = load_frag_B(W + 32, 128);
    v16h b2 = load_frag_B(W + 64, 128);
    v16h b3 = load_frag_B(W + 96, 128);
    for (int p = 0; p < 4; ++p) {
      const half_t* A = bufA + p * 2048;
      v8f acc = {0,0,0,0,0,0,0,0};
      acc = WMMA(load_frag_A(A +  0, 128), b0, acc);
      acc = WMMA(load_frag_A(A + 32, 128), b1, acc);
      acc = WMMA(load_frag_A(A + 64, 128), b2, acc);
      acc = WMMA(load_frag_A(A + 96, 128), b3, acc);
      store_act(acc, bias, nb, bufB + p * 2048, true);
    }
  }
  __syncthreads();

  // L4 (2x2 on 2x2 grid, K=512): act4 -> act5 (bufA) [16][128], relu
  {
    const half_t* W = wsH + OFF_W4 + g * 65536 + nb * 16 * 512;
    const float* bias = wsF + FB4 + g * 128;
    v8f acc = {0,0,0,0,0,0,0,0};
#pragma unroll
    for (int q = 0; q < 4; ++q)
#pragma unroll
      for (int kc = 0; kc < 4; ++kc) {
        v16h a = load_frag_A(bufB + q * 2048 + kc * 32, 128);
        v16h b = load_frag_B(W + (q * 4 + kc) * 32, 512);
        acc = WMMA(a, b, acc);
      }
    store_act(acc, bias, nb, bufA, true);
  }
  __syncthreads();

  // L5 (128->32, BN no relu) + clip[-1,QMAX] + quant(1/128) + atomic accumulate
  if (nb < 2) {
    const half_t* W = wsH + OFF_W5 + g * 4096 + nb * 16 * 128;
    const float* bias = wsF + FB5 + g * 32;
    v8f acc = {0,0,0,0,0,0,0,0};
#pragma unroll
    for (int ks = 0; ks < 4; ++ks) {
      v16h a = load_frag_A(bufA + ks * 32, 128);
      v16h b = load_frag_B(W + ks * 32, 128);
      acc = WMMA(a, b, acc);
    }
    int n = nb * 16 + (lane & 15);
    float bv = bias[n];
    int mb = (lane >> 4) * 8;
#pragma unroll
    for (int r = 0; r < 8; ++r) {
      float v = acc[r] + bv;
      v = fminf(fmaxf(v, -1.f), QMAXF);
      v = rintf(v * 128.f) * (1.f / 128.f);   // exact multiple of 1/128 -> f32 sums exact
      atomicAdd(&featAcc[(sBase + mb + r) * 32 + n], v);
    }
  }
}

// ---------------- quantized value MLP: 32->32->32->3 per batch row ----------------
__global__ __launch_bounds__(256) void mlp_kernel(
    const float* __restrict__ feat, const half_t* __restrict__ wsH,
    const float* __restrict__ wsF, float* __restrict__ out, int Bsz)
{
  __shared__ half_t X[8][16][32];
  const int wave = __builtin_amdgcn_readfirstlane(threadIdx.x) >> 5;
  const int lane = threadIdx.x & 31;
  const int m0 = (blockIdx.x * 8 + wave) * 16;
  half_t* Xw = &X[wave][0][0];

  // xq = round(clip(feature,-1,QMAX)*128)/128  -> exact in f16
  for (int idx = lane; idx < 512; idx += 32) {
    int r = idx >> 5, k = idx & 31;
    float v = feat[(m0 + r) * 32 + k];
    v = fminf(fmaxf(v, -1.f), QMAXF);
    v = rintf(v * 128.f) * (1.f / 128.f);
    Xw[idx] = (half_t)v;
  }

  const half_t* Ws[2] = { wsH + OFF_VW0, wsH + OFF_VW1 };
  const float*  Fs[2] = { wsF + FVB0,    wsF + FVB1 };
#pragma unroll
  for (int L = 0; L < 2; ++L) {
    v16h a = load_frag_A(Xw, 32);
    v16h b0 = load_frag_B(Ws[L], 32);
    v16h b1 = load_frag_B(Ws[L] + 16 * 32, 32);
    v8f acc0 = {0,0,0,0,0,0,0,0}, acc1 = {0,0,0,0,0,0,0,0};
    acc0 = WMMA(a, b0, acc0);
    acc1 = WMMA(a, b1, acc1);
    int n = lane & 15;
    half_t* dst = Xw + ((lane >> 4) * 8) * 32;
#pragma unroll
    for (int r = 0; r < 8; ++r) {
      float y0 = acc0[r] + Fs[L][n];
      y0 = fminf(fmaxf(y0, 0.f), QMAXF);
      dst[r * 32 + n] = (half_t)(rintf(y0 * 128.f) * (1.f / 128.f));
      float y1 = acc1[r] + Fs[L][16 + n];
      y1 = fminf(fmaxf(y1, 0.f), QMAXF);
      dst[r * 32 + 16 + n] = (half_t)(rintf(y1 * 128.f) * (1.f / 128.f));
    }
  }
  // final 32 -> 3 (N padded to 16)
  {
    v16h a = load_frag_A(Xw, 32);
    v16h b = load_frag_B(wsH + OFF_VW2, 32);
    v8f acc = {0,0,0,0,0,0,0,0};
    acc = WMMA(a, b, acc);
    int n = lane & 15;
    if (n < 3) {
      float bv = wsF[FVB2 + n];
      int mb = m0 + (lane >> 4) * 8;
#pragma unroll
      for (int r = 0; r < 8; ++r)
        out[(mb + r) * 3 + n] = acc[r] + bv;
    }
  }
  (void)Bsz;
}

extern "C" void kernel_launch(void* const* d_in, const int* in_sizes, int n_in,
                              void* d_out, int out_size, void* d_ws, size_t ws_size,
                              hipStream_t stream) {
  (void)n_in; (void)out_size; (void)ws_size;
  const float* x   = (const float*)d_in[0];
  const float* w0  = (const float*)d_in[1];
  const float* b0  = (const float*)d_in[2];
  const float* w1  = (const float*)d_in[3];
  const float* b1  = (const float*)d_in[4];
  const float* w2  = (const float*)d_in[5];
  const float* b2  = (const float*)d_in[6];
  const float* w3  = (const float*)d_in[7];
  const float* b3  = (const float*)d_in[8];
  const float* w4  = (const float*)d_in[9];
  const float* b4  = (const float*)d_in[10];
  const float* w5  = (const float*)d_in[11];
  const float* b5  = (const float*)d_in[12];
  const float* bn1 = (const float*)d_in[13];
  const float* bn2 = (const float*)d_in[14];
  const float* bn3 = (const float*)d_in[15];
  const float* bn4 = (const float*)d_in[16];
  const float* bn5 = (const float*)d_in[17];
  const float* vw0 = (const float*)d_in[18];
  const float* vb0 = (const float*)d_in[19];
  const float* vw1 = (const float*)d_in[20];
  const float* vb1 = (const float*)d_in[21];
  const float* vw2 = (const float*)d_in[22];
  const float* vb2 = (const float*)d_in[23];

  const int Bsz = in_sizes[0] / 98;            // x is [B][2][7][7]
  half_t* wsH = (half_t*)d_ws;
  float*  wsF = (float*)((char*)d_ws + (1 << 20));
  float*  featAcc = wsF + FFEAT;
  float*  out = (float*)d_out;

  hipMemsetAsync(featAcc, 0, (size_t)Bsz * 32 * sizeof(float), stream);
  hipMemsetAsync(out + (size_t)Bsz * 3, 0, (size_t)Bsz * 49 * sizeof(float), stream);

  prep_kernel<<<dim3(256), dim3(256), 0, stream>>>(
      w0, b0, w1, b1, w2, b2, w3, b3, w4, b4, w5, b5,
      bn1, bn2, bn3, bn4, bn5, vw0, vb0, vw1, vb1, vw2, vb2,
      wsH, wsF);

  // one workgroup per 16 patch-samples; 16 chunks * B samples / 16 = B workgroups
  conv_stack_kernel<<<dim3(Bsz), dim3(256), 73728, stream>>>(x, wsH, wsF, featAcc, Bsz);

  // 8 waves * 16 rows per block
  mlp_kernel<<<dim3(Bsz / 128), dim3(256), 0, stream>>>(featAcc, wsH, wsF, out, Bsz);
}
